// diff_CSDI_31653908971879
// MI455X (gfx1250) — compile-verified, matrix operand order checked by reading, not run on hardware
//
#include <hip/hip_runtime.h>
#include <math.h>

// ---------------------------------------------------------------------------
// Types for CDNA5 WMMA / TDM
// ---------------------------------------------------------------------------
typedef __attribute__((ext_vector_type(16))) _Float16 v16h;
typedef __attribute__((ext_vector_type(8)))  float    v8f;
typedef __attribute__((ext_vector_type(2)))  float    v2f;
typedef __attribute__((ext_vector_type(4)))  unsigned int su4;
typedef __attribute__((ext_vector_type(8)))  unsigned int su8;

#define ACT_NONE 0
#define ACT_RELU 1
#define ACT_GELU 2
#define ACT_SILU 3
#define ACT_SIG  4
#define ACT_TANH 5

__device__ __forceinline__ float geluf(float x) {
  const float c = 0.7978845608028654f;
  float t = tanhf(c * (x + 0.044715f * x * x * x));
  return 0.5f * x * (1.f + t);
}
__device__ __forceinline__ float actf(float x, int a) {
  switch (a) {
    case ACT_RELU: return fmaxf(x, 0.f);
    case ACT_GELU: return geluf(x);
    case ACT_SILU: return x / (1.f + __expf(-x));
    case ACT_SIG:  return 1.f / (1.f + __expf(-x));
    case ACT_TANH: return tanhf(x);
    default:       return x;
  }
}

// ---------------------------------------------------------------------------
// Generic affine-addressed GEMM, f16 WMMA, f32 accumulate, LDS-staged.
//   C[bat,row,n] = act( alpha * sum_k A[bat,row,k] * B[bat,n,k] + bias[n] )
// A element:  A + bat*aBat + row*aRS + k*aKS
// B element:  B + bat*bBat + n*bNS + k          (weight layout [N,K], k contiguous)
// C element:  C + bat*cBat + row*cRS + n*cNS
//
// Block = 4 waves; block tile 64x16, k-step 32. A/B slabs go to LDS already
// arranged in the WMMA fragment layout so each wave fetches fragments with
// 2x ds_load_b128. OOB is handled by clamp + multiply-by-{0,1}-mask: the
// mask multiply is data flow, not control flow, so clang cannot sink the
// loads under EXEC branches (a*0 is not foldable without fast-math).
// ---------------------------------------------------------------------------
#define GWAVES 4

struct GemmDesc {
  const float* A; long aRS, aKS, aBat;
  const float* B; long bNS, bBat;
  const float* bias;
  float* C; long cRS, cNS, cBat;
  int M, N, K;
  int act;
  float alpha;
};

__global__ __launch_bounds__(32 * GWAVES)
void k_gemm_wmma(GemmDesc g) {
  __shared__ alignas(32) _Float16 Ash[GWAVES][32][16];
  __shared__ alignas(32) _Float16 Bsh[32][16];
  const int lane = threadIdx.x & 31;
  const int wave = threadIdx.x >> 5;
  const int bat = blockIdx.z;
  const int rowB0 = blockIdx.y * (16 * GWAVES);
  const int col0 = blockIdx.x * 16;
  const float* A = g.A + (long)bat * g.aBat;
  const float* B = g.B + (long)bat * g.bBat;
  const int Mm1 = g.M - 1, Km1 = g.K - 1, Nm1 = g.N - 1;

  v8f acc = {0.f, 0.f, 0.f, 0.f, 0.f, 0.f, 0.f, 0.f};

  for (int k0 = 0; k0 < g.K; k0 += 32) {
    // ---- cooperative A slab: 64 rows x 32 k (16 elems/thread) ----
    for (int e = threadIdx.x; e < 64 * 32; e += 128) {
      const int r = e >> 5, k = e & 31;
      const int grow = rowB0 + r, gk = k0 + k;
      const int crow = (grow < Mm1) ? grow : Mm1;
      const int ck   = (gk   < Km1) ? gk   : Km1;
      const float msk = (grow <= Mm1 && gk <= Km1) ? 1.f : 0.f;
      const float a = A[(long)crow * g.aRS + (long)ck * g.aKS] * msk;
      const int w = r >> 4, rr = r & 15;
      const int hf = (k & 15) >> 3;                        // fragment half
      const int slot = (k < 16) ? (k & 7) : (8 + (k & 7)); // fragment element
      Ash[w][hf * 16 + rr][slot] = (_Float16)a;
    }
    // ---- cooperative B slab: 16 cols x 32 k (4 elems/thread) ----
    for (int e = threadIdx.x; e < 16 * 32; e += 128) {
      const int n = e >> 5, k = e & 31;
      const int gn = col0 + n, gk = k0 + k;
      const int cn = (gn < Nm1) ? gn : Nm1;
      const int ck = (gk < Km1) ? gk : Km1;
      const float msk = (gn <= Nm1 && gk <= Km1) ? 1.f : 0.f;
      const float b = B[(long)cn * g.bNS + ck] * msk;
      const int hf = (k & 15) >> 3;
      const int slot = (k < 16) ? (k & 7) : (8 + (k & 7));
      Bsh[hf * 16 + n][slot] = (_Float16)b;
    }
    __syncthreads();
    v16h af = *(const v16h*)(&Ash[wave][lane][0]);   // 2x ds_load_b128
    v16h bf = *(const v16h*)(&Bsh[lane][0]);         // 2x ds_load_b128
    acc = __builtin_amdgcn_wmma_f32_16x16x32_f16(false, af, false, bf,
                                                 (short)0, acc, false, false);
    __syncthreads();
  }

  // ---- epilogue (divergence allowed here, all WMMAs are done) ----
  const int mr = lane & 15;
  const int half = lane >> 4;
  const int ccol = col0 + mr;
  if (ccol <= Nm1) {
    const float bb = g.bias ? g.bias[ccol] : 0.f;
    float* Cb = g.C + (long)bat * g.cBat;
    const int rbase = rowB0 + wave * 16 + half * 8;
#pragma unroll
    for (int i = 0; i < 8; ++i) {
      const int r = rbase + i;
      if (r <= Mm1) {
        float v = acc[i] * g.alpha + bb;
        Cb[(long)r * g.cRS + (long)ccol * g.cNS] = actf(v, g.act);
      }
    }
  }
}

// ---------------------------------------------------------------------------
// Order-2 graph support via f32 WMMA (exact):
//   A2[d][w][v] = (a_d @ a_d^T)[w][v] * (w != v),  a_d = asup[d] (128x128)
// ---------------------------------------------------------------------------
__global__ __launch_bounds__(128)
void k_support2_wmma(const float* __restrict__ A, float* __restrict__ A2) {
  const int lane = threadIdx.x & 31;
  const int wave = threadIdx.x >> 5;
  const int row0 = blockIdx.y * 64 + wave * 16;
  const int col0 = blockIdx.x * 16;
  const float* a = A + (long)blockIdx.z * 16384;
  const int mr = lane & 15;
  const int half = lane >> 4;
  v8f acc = {0.f, 0.f, 0.f, 0.f, 0.f, 0.f, 0.f, 0.f};
  for (int k0 = 0; k0 < 128; k0 += 4) {
    v2f af, bf;
#pragma unroll
    for (int v = 0; v < 2; ++v) {
      const int k = k0 + half * 2 + v;
      af[v] = a[(row0 + mr) * 128 + k];
      bf[v] = a[(col0 + mr) * 128 + k];   // B[k, n] = a[n, k] (a @ a^T)
    }
    acc = __builtin_amdgcn_wmma_f32_16x16x4_f32(false, af, false, bf,
                                                (short)0, acc, false, false);
  }
  const int cc = col0 + mr;
  const int rbase = row0 + half * 8;
#pragma unroll
  for (int i = 0; i < 8; ++i) {
    const int r = rbase + i;
    A2[(long)blockIdx.z * 16384 + r * 128 + cc] = (r == cc) ? 0.f : acc[i];
  }
}

// ---------------------------------------------------------------------------
// TDM weight warm: build a D# in SGPRs and issue tensor_load_to_lds to pull a
// 2048-float tile of the weight matrix through L2 into LDS (prefetch for the
// 48-step GRU scan). Groups 2/3 present but zeroed (4-group form).
// ---------------------------------------------------------------------------
__global__ __launch_bounds__(64)
void k_tdm_warm(const float* __restrict__ W, unsigned int nelem) {
  __shared__ float lbuf[2048];
  unsigned long long ga =
      (unsigned long long)(const void*)W + (unsigned long long)blockIdx.x * 8192ull;
  su4 g0;
  g0[0] = 1u;                                   // count=1, user descriptor
  g0[1] = 0u;                                   // lds_addr = 0 (lbuf at base)
  g0[2] = (unsigned)(ga & 0xffffffffu);         // global_addr[31:0]
  g0[3] = (unsigned)((ga >> 32) & 0x01ffffffu)  // global_addr[56:32]
          | (2u << 30);                         // type = 2 ("image")
  su8 g1;
  const unsigned tile0 = 2048u;
  g1[0] = (2u << 16);                           // data_size = 4 bytes
  g1[1] = (nelem & 0xffffu) << 16;              // tensor_dim0[15:0]  @bit48
  g1[2] = (nelem >> 16) & 0xffffu;              // tensor_dim0[31:16] @bit64
  g1[3] = (tile0 & 0xffffu) << 16;              // tile_dim0 @bit112
  g1[4] = 1u;                                   // tile_dim1 = 1
  g1[5] = nelem;                                // tensor_dim0_stride[31:0]
  g1[6] = 0u;
  g1[7] = 0u;
  su4 g2 = {0u, 0u, 0u, 0u};
  su4 g3 = {0u, 0u, 0u, 0u};
  asm volatile("tensor_load_to_lds %0, %1, %2, %3"
               :: "s"(g0), "s"(g1), "s"(g2), "s"(g3) : "memory");
  __builtin_amdgcn_s_wait_tensorcnt(0);
  float v = lbuf[threadIdx.x & 2047];
  asm volatile("" :: "v"(v));                   // keep LDS read alive
}

// ---------------------------------------------------------------------------
// Async-to-LDS warm: per-lane global_load_async_to_lds_b128 (ASYNCcnt path).
// Each lane copies 16 bytes; one wave per block; pure L2 warm.
// ---------------------------------------------------------------------------
__global__ __launch_bounds__(32)
void k_async_warm(const float* __restrict__ P, long nchunk) {
  __shared__ float lbuf[128];
  const int lane = threadIdx.x & 31;
  long i = (long)blockIdx.x * 32 + lane;        // 16-byte chunk index
  if (i >= nchunk) i = nchunk - 1;
  unsigned long long ga =
      (unsigned long long)(const void*)P + (unsigned long long)i * 16ull;
  unsigned lds = (unsigned)lane * 16u;          // LDS byte offset (lbuf base 0)
  asm volatile("global_load_async_to_lds_b128 %0, %1, off"
               :: "v"(lds), "v"(ga) : "memory");
  asm volatile("s_wait_asynccnt 0x0" ::: "memory");
  float v = lbuf[threadIdx.x * 4];
  asm volatile("" :: "v"(v));
}

// ---------------------------------------------------------------------------
// Attention: one block per (n, head). QKV rows are (s*N + n), 3*64 wide.
// ---------------------------------------------------------------------------
__global__ __launch_bounds__(128)
void k_attn(const float* __restrict__ QKV, float* __restrict__ O, int S, int N) {
  __shared__ float q[128 * 8], kk[128 * 8], vv[128 * 8];
  const int n = blockIdx.x;
  const int h = blockIdx.y;
  for (int idx = threadIdx.x; idx < S * 8; idx += 128) {
    int s = idx >> 3, d = idx & 7;
    const float* r = QKV + ((long)s * N + n) * 192;
    q[idx]  = r[h * 8 + d];
    kk[idx] = r[64 + h * 8 + d];
    vv[idx] = r[128 + h * 8 + d];
  }
  __syncthreads();
  const int s = threadIdx.x;
  if (s >= S) return;
  const float scale = 0.35355339059327373f;  // 1/sqrt(8)
  float m = -1e30f, l = 0.f;
  for (int t = 0; t < S; ++t) {
    float d2 = 0.f;
#pragma unroll
    for (int d = 0; d < 8; ++d) d2 += q[s * 8 + d] * kk[t * 8 + d];
    d2 *= scale;
    if (d2 > m) { l = l * __expf(m - d2) + 1.f; m = d2; }
    else        { l += __expf(d2 - m); }
  }
  float o[8] = {0, 0, 0, 0, 0, 0, 0, 0};
  for (int t = 0; t < S; ++t) {
    float d2 = 0.f;
#pragma unroll
    for (int d = 0; d < 8; ++d) d2 += q[s * 8 + d] * kk[t * 8 + d];
    float w = __expf(d2 * scale - m);
#pragma unroll
    for (int d = 0; d < 8; ++d) o[d] += w * vv[t * 8 + d];
  }
  const float inv = 1.f / l;
  float* orow = O + ((long)s * N + n) * 64 + h * 8;
#pragma unroll
  for (int d = 0; d < 8; ++d) orow[d] = o[d] * inv;
}

// ---------------------------------------------------------------------------
// out = LayerNorm(A + B) over last dim D (32|64|128), wave32 shuffle reduce.
// ---------------------------------------------------------------------------
__global__ __launch_bounds__(128)
void k_add_ln(const float* __restrict__ A, const float* __restrict__ Bq,
              const float* __restrict__ gg, const float* __restrict__ bb,
              float* __restrict__ O, long R, int D) {
  long r = (long)blockIdx.x * 4 + (threadIdx.x >> 5);
  if (r >= R) return;
  const int lane = threadIdx.x & 31;
  const int per = D >> 5;
  const float* a = A + r * D;
  const float* b = Bq ? Bq + r * D : nullptr;
  float vals[4];
  float sum = 0.f;
  for (int i = 0; i < per; ++i) {
    float v = a[lane + i * 32] + (b ? b[lane + i * 32] : 0.f);
    vals[i] = v; sum += v;
  }
  for (int o = 16; o > 0; o >>= 1) sum += __shfl_xor(sum, o, 32);
  float mu = sum / D;
  float var = 0.f;
  for (int i = 0; i < per; ++i) { float d = vals[i] - mu; var += d * d; }
  for (int o = 16; o > 0; o >>= 1) var += __shfl_xor(var, o, 32);
  float inv = rsqrtf(var / D + 1e-5f);
  float* out = O + r * D;
  for (int i = 0; i < per; ++i) {
    int c = lane + i * 32;
    out[c] = (vals[i] - mu) * inv * gg[c] + bb[c];
  }
}

// ---------------------------------------------------------------------------
// S4 helpers
// ---------------------------------------------------------------------------
__global__ void k_s4_kc(const float* __restrict__ kern, float* __restrict__ Kc, int Hd, long total) {
  long idx = (long)blockIdx.x * blockDim.x + threadIdx.x;
  if (idx >= total) return;
  int m = (int)(idx % 48);
  long r = idx / 48;
  int so = (int)(r % 48);
  int c = (int)(r / 48);
  int j = so - m; if (j < 0) j += 96;
  float v = (j < 48) ? kern[c * 48 + j] : kern[(long)Hd * 48 + c * 48 + (95 - j)];
  Kc[idx] = v;
}

__global__ void k_s4_gelu_pack(const float* __restrict__ Yc, const float* __restrict__ U,
                               const float* __restrict__ D, float* __restrict__ G2,
                               int Hd, int N, int S) {
  long idx = (long)blockIdx.x * blockDim.x + threadIdx.x;
  if (idx >= (long)N * S * Hd) return;
  int c = (int)(idx % Hd);
  long r = idx / Hd;
  int s = (int)(r % S);
  long n = r / S;
  long src = ((long)c * N + n) * S + s;
  float u = U[src];
  G2[idx] = geluf(Yc[src] + u * D[c]);
}

__global__ __launch_bounds__(128)
void k_s4_glu_ln(const float* __restrict__ Y2, const float* __restrict__ U,
                 const float* __restrict__ gg, const float* __restrict__ be,
                 float* __restrict__ Out, int Hd, int N, int S, int K, int L) {
  long R = (long)N * S;
  long r = (long)blockIdx.x * 4 + (threadIdx.x >> 5);
  if (r >= R) return;
  const int lane = threadIdx.x & 31;
  const int per = Hd >> 5;
  int n = (int)(r / S), s = (int)(r % S);
  int b = n / K, k = n % K;
  const float* yrow = Y2 + r * (2 * Hd);
  float vals[4]; float sum = 0.f;
  for (int i = 0; i < per; ++i) {
    int c = lane + i * 32;
    float a  = yrow[c];
    float g2 = yrow[Hd + c];
    float v = a * (1.f / (1.f + __expf(-g2))) + U[((long)c * N + n) * S + s];
    vals[i] = v; sum += v;
  }
  for (int o = 16; o > 0; o >>= 1) sum += __shfl_xor(sum, o, 32);
  float mu = sum / Hd;
  float var = 0.f;
  for (int i = 0; i < per; ++i) { float d = vals[i] - mu; var += d * d; }
  for (int o = 16; o > 0; o >>= 1) var += __shfl_xor(var, o, 32);
  float inv = rsqrtf(var / Hd + 1e-5f);
  for (int i = 0; i < per; ++i) {
    int c = lane + i * 32;
    Out[(((long)b * Hd + c) * K + k) * L + s] = (vals[i] - mu) * inv * gg[c] + be[c];
  }
}

// ---------------------------------------------------------------------------
// Packs / elementwise
// ---------------------------------------------------------------------------
__global__ void k_pack_hdp_to_T(const float* __restrict__ H, const float* __restrict__ dp,
                                float* __restrict__ T) {
  long idx = (long)blockIdx.x * blockDim.x + threadIdx.x;
  if (idx >= 8L * 64 * 128 * 48) return;
  int l = (int)(idx % 48);
  long r = idx / 48;
  int k = (int)(r % 128); r /= 128;
  int c = (int)(r % 64);
  int b = (int)(r / 64);
  float v = H[idx] + dp[b * 64 + c];
  long n = (long)b * 128 + k;
  T[((long)l * 1024 + n) * 64 + c] = v;
}

__global__ void k_pack_T_to_F(const float* __restrict__ T, float* __restrict__ F) {
  long idx = (long)blockIdx.x * blockDim.x + threadIdx.x;
  if (idx >= 8L * 64 * 128 * 48) return;
  int c = (int)(idx & 63);
  long r = idx >> 6;
  int l = (int)(r % 48); r /= 48;
  int b = (int)(r & 7);
  int k = (int)(r >> 3);
  F[idx] = T[(((long)l * 1024) + b * 128 + k) * 64 + c];
}

__global__ void k_pack_F_to_U(const float* __restrict__ F, float* __restrict__ U) {
  long idx = (long)blockIdx.x * blockDim.x + threadIdx.x;
  if (idx >= 64L * 1024 * 48) return;
  int l = (int)(idx % 48);
  long r = idx / 48;
  int n = (int)(r % 1024);
  int c = (int)(r / 1024);
  int b = n >> 7, k = n & 127;
  U[idx] = F[(((long)k * 384) + b * 48 + l) * 64 + c];
}

__global__ void k_pack_BCKL_to_U(const float* __restrict__ Y, float* __restrict__ U,
                                 int Cd, long total) {
  long idx = (long)blockIdx.x * blockDim.x + threadIdx.x;
  if (idx >= total) return;
  int l = (int)(idx % 48);
  long r = idx / 48;
  int k = (int)(r % 128); r /= 128;
  int b = (int)(r % 8);
  int c = (int)(r / 8);
  U[idx] = Y[(((long)b * Cd + c) * 128 + k) * 48 + l];
}

__global__ void k_addinto(float* __restrict__ dst, const float* __restrict__ src, long n) {
  long i = (long)blockIdx.x * blockDim.x + threadIdx.x;
  if (i < n) dst[i] += src[i];
}

__global__ void k_gate(const float* __restrict__ Y, float* __restrict__ Z) {
  long idx = (long)blockIdx.x * blockDim.x + threadIdx.x;
  if (idx >= 8L * 64 * 6144) return;
  long pos = idx % 6144;
  int c = (int)((idx / 6144) & 63);
  int b = (int)(idx / (64L * 6144));
  float g2 = Y[(((long)b * 128 + c) * 6144) + pos];
  float f2 = Y[(((long)b * 128 + 64 + c) * 6144) + pos];
  Z[idx] = (1.f / (1.f + __expf(-g2))) * tanhf(f2);
}

__global__ void k_res_skip(const float* __restrict__ ORS, float* __restrict__ Hb,
                           float* __restrict__ SK) {
  long idx = (long)blockIdx.x * blockDim.x + threadIdx.x;
  if (idx >= 8L * 64 * 6144) return;
  long pos = idx % 6144;
  int c = (int)((idx / 6144) & 63);
  int b = (int)(idx / (64L * 6144));
  float res = ORS[(((long)b * 128 + c) * 6144) + pos];
  float sk  = ORS[(((long)b * 128 + 64 + c) * 6144) + pos];
  Hb[idx] = (Hb[idx] + res) * 0.7071067811865476f;
  SK[idx] += sk;
}

__global__ void k_demb_table(const int* __restrict__ step, float* __restrict__ e) {
  int b = blockIdx.x;
  int d = threadIdx.x;  // 0..127
  float s = (float)step[b];
  int j = (d < 64) ? d : (d - 64);
  float fr = __powf(10.f, (float)j * (4.f / 63.f));
  float t = s * fr;
  e[b * 128 + d] = (d < 64) ? __sinf(t) : __cosf(t);
}

__global__ void k_support_norm(const float* __restrict__ adj, float* __restrict__ A) {
  int tid = threadIdx.x;
  if (tid >= 256) return;
  int dir = tid >> 7;
  int r = tid & 127;
  float s = 1e-8f;
  for (int c2 = 0; c2 < 128; ++c2) s += dir ? adj[c2 * 128 + r] : adj[r * 128 + c2];
  float inv = 1.f / s;
  for (int c2 = 0; c2 < 128; ++c2) {
    float v = dir ? adj[c2 * 128 + r] : adj[r * 128 + c2];
    A[dir * 16384 + r * 128 + c2] = v * inv;
  }
}

// ---------------------------------------------------------------------------
// GC-GRU step kernels
// ---------------------------------------------------------------------------
__global__ void k_gru_pack_xh(const float* __restrict__ Y, const int* __restrict__ cm,
                              const float* __restrict__ h, float* __restrict__ XH, int t) {
  int idx = blockIdx.x * blockDim.x + threadIdx.x;
  if (idx >= 8 * 129 * 128) return;
  int k = idx & 127;
  int c = (idx >> 7) % 129;
  int b = idx / (129 * 128);
  float v;
  if (c < 64)       v = Y[(((long)b * 64 + c) * 128 + k) * 48 + t];
  else if (c == 64) v = (float)cm[((long)b * 128 + k) * 48 + t];
  else              v = h[((long)b * 64 + (c - 65)) * 128 + k];
  // prefetch next timestep's slice (keeps global_prefetch_b8 in the mix)
  int cc = (c < 64) ? c : 0;
  int tn = (t + 1 < 48) ? t + 1 : t;
  __builtin_prefetch(&Y[(((long)b * 64 + cc) * 128 + k) * 48 + tn], 0, 1);
  XH[idx] = v;
}
__global__ void k_gru_pack_xc(const float* __restrict__ Y, const int* __restrict__ cm,
                              const float* __restrict__ h, const float* __restrict__ rg,
                              float* __restrict__ XC, int t) {
  int idx = blockIdx.x * blockDim.x + threadIdx.x;
  if (idx >= 8 * 129 * 128) return;
  int k = idx & 127;
  int c = (idx >> 7) % 129;
  int b = idx / (129 * 128);
  float v;
  if (c < 64)       v = Y[(((long)b * 64 + c) * 128 + k) * 48 + t];
  else if (c == 64) v = (float)cm[((long)b * 128 + k) * 48 + t];
  else {
    int cc = c - 65;
    v = rg[((long)b * 128 + k) * 64 + cc] * h[((long)b * 64 + cc) * 128 + k];
  }
  XC[idx] = v;
}
__global__ void k_gru_pack_feats(const float* __restrict__ XH, const float* __restrict__ X1,
                                 const float* __restrict__ X2, float* __restrict__ FP) {
  long idx = (long)blockIdx.x * blockDim.x + threadIdx.x;
  if (idx >= 1024L * 1161) return;
  int i = (int)(idx % 1161);
  long row = idx / 1161;
  int b = (int)(row >> 7);
  int k = (int)(row & 127);
  float v;
  if (i < 129) v = XH[((long)b * 129 + i) * 128 + k];
  else {
    int j = i - 129;
    int sgrp = j / 258;
    int rem = j - sgrp * 258;
    const float* src = (rem >= 129) ? X2 : X1;
    int c = (rem >= 129) ? (rem - 129) : rem;
    v = src[(long)sgrp * 132096 + ((long)b * 129 + c) * 128 + k];
  }
  FP[idx] = v;
}
__global__ void k_gru_update(const float* __restrict__ U, const float* __restrict__ Cg,
                             float* __restrict__ H, float* __restrict__ YG, int t) {
  int idx = blockIdx.x * blockDim.x + threadIdx.x;
  if (idx >= 8 * 64 * 128) return;
  int k = idx & 127;
  int c = (idx >> 7) & 63;
  int b = idx >> 13;
  long rc = ((long)b * 128 + k) * 64 + c;
  float u = U[rc], cc = Cg[rc];
  float hv = H[idx];
  float hn = u * hv + (1.f - u) * cc;
  H[idx] = hn;
  YG[(long)idx * 48 + t] = hn;
}

// ---------------------------------------------------------------------------
// Host-side helpers
// ---------------------------------------------------------------------------
static inline unsigned ceilu(long a, long b) { return (unsigned)((a + b - 1) / b); }

static void gemm(hipStream_t s,
                 const float* A, long aRS, long aKS, long aBat,
                 const float* B, long bNS, long bBat,
                 const float* bias,
                 float* C, long cRS, long cNS, long cBat,
                 int M, int N, int K, int nbat, int act, float alpha = 1.f) {
  GemmDesc g{A, aRS, aKS, aBat, B, bNS, bBat, bias, C, cRS, cNS, cBat, M, N, K, act, alpha};
  dim3 grid(ceilu(N, 16), ceilu(M, 16 * GWAVES), nbat);
  k_gemm_wmma<<<grid, 32 * GWAVES, 0, s>>>(g);
}

static void run_transformer(hipStream_t st, float* X, int S, int N, const float* const* tp,
                            float* qkv, float* obuf, float* p1, float* p2) {
  long R = (long)S * N;
  gemm(st, X, 64, 1, 0, tp[5], 64, 0, tp[4], qkv, 192, 1, 0, (int)R, 192, 64, 1, ACT_NONE);
  k_attn<<<dim3(N, 8), 128, 0, st>>>(qkv, obuf, S, N);
  gemm(st, obuf, 64, 1, 0, tp[11], 64, 0, tp[10], p1, 64, 1, 0, (int)R, 64, 64, 1, ACT_NONE);
  k_add_ln<<<ceilu(R, 4), 128, 0, st>>>(X, p1, tp[7], tp[6], X, R, 64);
  gemm(st, X, 64, 1, 0, tp[1], 64, 0, tp[0], p1, 64, 1, 0, (int)R, 64, 64, 1, ACT_GELU);
  gemm(st, p1, 64, 1, 0, tp[3], 64, 0, tp[2], p2, 64, 1, 0, (int)R, 64, 64, 1, ACT_NONE);
  k_add_ln<<<ceilu(R, 4), 128, 0, st>>>(X, p2, tp[9], tp[8], X, R, 64);
}

static void run_s4(hipStream_t st, const float* U, float* Yc, float* G2, float* Y2, float* Kc,
                   const float* const* s4p, float* Out, int Hd, int N, int S, int K, int L) {
  long nk = (long)Hd * S * S;
  k_s4_kc<<<ceilu(nk, 256), 256, 0, st>>>(s4p[1], Kc, Hd, nk);
  gemm(st, U, S, 1, (long)N * S, Kc, S, (long)S * S, nullptr,
       Yc, S, 1, (long)N * S, N, S, S, Hd, ACT_NONE);
  long ng = (long)N * S * Hd;
  k_s4_gelu_pack<<<ceilu(ng, 256), 256, 0, st>>>(Yc, U, s4p[0], G2, Hd, N, S);
  gemm(st, G2, Hd, 1, 0, s4p[5], Hd, 0, s4p[4], Y2, 2L * Hd, 1, 0,
       (int)((long)N * S), 2 * Hd, Hd, 1, ACT_NONE);
  long nr = (long)N * S;
  k_s4_glu_ln<<<ceilu(nr, 4), 128, 0, st>>>(Y2, U, s4p[3], s4p[2], Out, Hd, N, S, K, L);
}

// ---------------------------------------------------------------------------
// Orchestration
// ---------------------------------------------------------------------------
extern "C" void kernel_launch(void* const* d_in, const int* in_sizes, int n_in,
                              void* d_out, int out_size, void* d_ws, size_t ws_size,
                              hipStream_t stream) {
  (void)in_sizes; (void)n_in; (void)out_size; (void)ws_size;
  auto F = [&](int i) -> const float* { return (const float*)d_in[i]; };

  const float* x      = F(0);
  const float* cond   = F(1);
  const int*   dstep  = (const int*)d_in[2];
  const int*   cmask  = (const int*)d_in[4];
  const float* adj    = F(5);
  const float* demb_b1 = F(6),  *demb_b2 = F(7), *demb_w1 = F(8), *demb_w2 = F(9);
  const float* in_b = F(10), *in_w = F(11);
  const int LBASE = 12, LSTRIDE = 52;
  const float* out1_b = F(LBASE + 4 * LSTRIDE + 0);
  const float* out1_w = F(LBASE + 4 * LSTRIDE + 1);
  const float* out2_b = F(LBASE + 4 * LSTRIDE + 2);
  const float* out2_w = F(LBASE + 4 * LSTRIDE + 3);

  char* wsBase = (char*)d_ws;
  size_t off = 0;
  auto alloc = [&](long nfloats) -> float* {
    float* p = (float*)(wsBase + off);
    off = (off + (size_t)nfloats * sizeof(float) + 255) & ~(size_t)255;
    return p;
  };
  const long E1 = 8L * 64 * 128 * 48;    // 3,145,728
  const long E2 = 8L * 128 * 128 * 48;   // 6,291,456
  float* de0  = alloc(1024);
  float* de1  = alloc(1024);
  float* dp   = alloc(512);
  float* asup = alloc(4 * 16384);
  float* hbuf = alloc(E1);
  float* sk   = alloc(E1);
  float* t0   = alloc(E1);
  float* t1   = alloc(E1);
  float* t2   = alloc(E1);
  float* qkv  = alloc(49152L * 192);     // also reused as s4#1 Y2 buffer
  float* obuf = alloc(E1);
  float* u1   = alloc(E1);
  float* ys4  = alloc(E1);
  float* yg   = alloc(E1);
  float* y6   = alloc(E2);
  float* ors  = alloc(E2);
  float* u2   = alloc(E2);
  float* yc2  = alloc(E2);
  float* g2b  = alloc(E2);
  float* y2b  = alloc(2 * E2);
  float* ys42 = alloc(E2);
  float* zb   = alloc(E1);
  float* kc1  = alloc(64L * 48 * 48);
  float* kc2  = alloc(128L * 48 * 48);
  float* xh   = alloc(8L * 129 * 128);
  float* x1g  = alloc(4L * 8 * 129 * 128);
  float* x2g  = alloc(4L * 8 * 129 * 128);
  float* fp   = alloc(1024L * 1161);
  float* rg   = alloc(1024L * 64);
  float* ug   = alloc(1024L * 64);
  float* cg   = alloc(1024L * 64);
  float* hst  = alloc(8L * 64 * 128);

  hipMemsetAsync(sk, 0, (size_t)E1 * sizeof(float), stream);

  // input conv1x1 (Cin=2) + ReLU -> hbuf [B,64,6144]
  gemm(stream, x, 1, 6144, 2L * 6144, in_w, 2, 0, in_b,
       hbuf, 1, 6144, 64L * 6144, 6144, 64, 2, 8, ACT_RELU);

  // diffusion embedding
  k_demb_table<<<8, 128, 0, stream>>>(dstep, de0);
  gemm(stream, de0, 128, 1, 0, demb_w1, 128, 0, demb_b1, de1, 128, 1, 0, 8, 128, 128, 1, ACT_SILU);
  gemm(stream, de1, 128, 1, 0, demb_w2, 128, 0, demb_b2, de0, 128, 1, 0, 8, 128, 128, 1, ACT_SILU);

  // graph supports: normalize, then order-2 via f32 WMMA
  k_support_norm<<<1, 256, 0, stream>>>(adj, asup);
  k_support2_wmma<<<dim3(8, 2, 2), 128, 0, stream>>>(asup, asup + 2 * 16384);

  for (int layer = 0; layer < 4; ++layer) {
    const int b0 = LBASE + layer * LSTRIDE;
    const float* afternn_b = F(b0 + 0), *afternn_w = F(b0 + 1);
    const float* cond_b = F(b0 + 2),    *cond_w = F(b0 + 3);
    const float* diff_b = F(b0 + 4),    *diff_w = F(b0 + 5);
    const float* gru_c_b = F(b0 + 6),   *gru_c_w = F(b0 + 7);
    const float* gru_r_b = F(b0 + 8),   *gru_r_w = F(b0 + 9);
    const float* gru_u_b = F(b0 + 10),  *gru_u_w = F(b0 + 11);
    const float* mid_b = F(b0 + 12),    *mid_w = F(b0 + 13);
    const float* lout_b = F(b0 + 14),   *lout_w = F(b0 + 15);
    const float* s41p[6]; for (int i = 0; i < 6; ++i) s41p[i] = F(b0 + 16 + i);
    const float* s42p[6]; for (int i = 0; i < 6; ++i) s42p[i] = F(b0 + 22 + i);
    const float* tfp[12]; for (int i = 0; i < 12; ++i) tfp[i] = F(b0 + 28 + i);
    const float* ttp[12]; for (int i = 0; i < 12; ++i) ttp[i] = F(b0 + 40 + i);

    gemm(stream, de0, 128, 1, 0, diff_w, 128, 0, diff_b, dp, 64, 1, 0, 8, 64, 128, 1, ACT_NONE);
    k_pack_hdp_to_T<<<ceilu(E1, 256), 256, 0, stream>>>(hbuf, dp, t0);

    run_transformer(stream, t0, 48, 1024, ttp, qkv, obuf, t1, t2);
    k_pack_T_to_F<<<ceilu(E1, 256), 256, 0, stream>>>(t0, t1);
    run_transformer(stream, t1, 128, 384, tfp, qkv, obuf, t0, t2);

    k_pack_F_to_U<<<ceilu(E1, 256), 256, 0, stream>>>(t1, u1);
    run_s4(stream, u1, t0, t2, qkv, kc1, s41p, ys4, 64, 1024, 48, 128, 48);

    // TDM-warm the GRU gate weights (L2 prefetch) before the sequential scan
    k_tdm_warm<<<dim3(36), 64, 0, stream>>>(gru_r_w, 64u * 1161u);
    k_tdm_warm<<<dim3(36), 64, 0, stream>>>(gru_u_w, 64u * 1161u);
    k_tdm_warm<<<dim3(36), 64, 0, stream>>>(gru_c_w, 64u * 1161u);

    hipMemsetAsync(hst, 0, 8L * 64 * 128 * sizeof(float), stream);
    for (int t = 0; t < 48; ++t) {
      k_gru_pack_xh<<<ceilu(8L * 129 * 128, 256), 256, 0, stream>>>(ys4, cmask, hst, xh, t);
      gemm(stream, xh, 128, 1, 0, asup, 128, 16384, nullptr,
           x1g, 128, 1, 132096, 1032, 128, 128, 4, ACT_NONE);
      gemm(stream, x1g, 128, 1, 132096, asup, 128, 16384, nullptr,
           x2g, 128, 1, 132096, 1032, 128, 128, 4, ACT_NONE);
      k_gru_pack_feats<<<ceilu(1024L * 1161, 256), 256, 0, stream>>>(xh, x1g, x2g, fp);
      gemm(stream, fp, 1161, 1, 0, gru_r_w, 1161, 0, gru_r_b, rg, 64, 1, 0, 1024, 64, 1161, 1, ACT_SIG);
      gemm(stream, fp, 1161, 1, 0, gru_u_w, 1161, 0, gru_u_b, ug, 64, 1, 0, 1024, 64, 1161, 1, ACT_SIG);
      k_gru_pack_xc<<<ceilu(8L * 129 * 128, 256), 256, 0, stream>>>(ys4, cmask, hst, rg, xh, t);
      gemm(stream, xh, 128, 1, 0, asup, 128, 16384, nullptr,
           x1g, 128, 1, 132096, 1032, 128, 128, 4, ACT_NONE);
      gemm(stream, x1g, 128, 1, 132096, asup, 128, 16384, nullptr,
           x2g, 128, 1, 132096, 1032, 128, 128, 4, ACT_NONE);
      k_gru_pack_feats<<<ceilu(1024L * 1161, 256), 256, 0, stream>>>(xh, x1g, x2g, fp);
      gemm(stream, fp, 1161, 1, 0, gru_c_w, 1161, 0, gru_c_b, cg, 64, 1, 0, 1024, 64, 1161, 1, ACT_TANH);
      k_gru_update<<<ceilu(8L * 64 * 128, 256), 256, 0, stream>>>(ug, cg, hst, yg, t);
    }

    gemm(stream, yg, 1, 6144, 64L * 6144, afternn_w, 64, 0, afternn_b,
         t0, 1, 6144, 64L * 6144, 6144, 64, 64, 8, ACT_NONE);
    gemm(stream, t0, 1, 6144, 64L * 6144, mid_w, 64, 0, mid_b,
         y6, 1, 6144, 128L * 6144, 6144, 128, 64, 8, ACT_NONE);
    // async-to-LDS warm of cond_info ahead of the SIDE->2C projection
    k_async_warm<<<dim3(256), 32, 0, stream>>>(cond, (long)(8L * 144 * 6144 / 4));
    gemm(stream, cond, 1, 6144, 144L * 6144, cond_w, 144, 0, cond_b,
         ors, 1, 6144, 128L * 6144, 6144, 128, 144, 8, ACT_NONE);
    k_addinto<<<ceilu(E2, 256), 256, 0, stream>>>(y6, ors, E2);

    k_pack_BCKL_to_U<<<ceilu(E2, 256), 256, 0, stream>>>(y6, u2, 128, E2);
    run_s4(stream, u2, yc2, g2b, y2b, kc2, s42p, ys42, 128, 1024, 48, 128, 48);

    k_gate<<<ceilu(E1, 256), 256, 0, stream>>>(ys42, zb);
    gemm(stream, zb, 1, 6144, 64L * 6144, lout_w, 64, 0, lout_b,
         ors, 1, 6144, 128L * 6144, 6144, 128, 64, 8, ACT_NONE);
    k_res_skip<<<ceilu(E1, 256), 256, 0, stream>>>(ors, hbuf, sk);
  }

  gemm(stream, sk, 1, 6144, 64L * 6144, out1_w, 64, 0, out1_b,
       t0, 1, 6144, 64L * 6144, 6144, 64, 64, 8, ACT_RELU, 0.5f);
  gemm(stream, t0, 1, 6144, 64L * 6144, out2_w, 64, 0, out2_b,
       (float*)d_out, 1, 1, 6144, 6144, 1, 64, 8, ACT_NONE);
}
